// QuantizedLinear_22634477650432
// MI455X (gfx1250) — compile-verified, matrix operand order checked by reading, not run on hardware
//
#include <hip/hip_runtime.h>

typedef __attribute__((ext_vector_type(16))) _Float16 v16h;
typedef __attribute__((ext_vector_type(8)))  _Float16 v8h;
typedef __attribute__((ext_vector_type(8)))  float    v8f;
typedef __attribute__((ext_vector_type(4)))  float    v4f;
typedef __attribute__((ext_vector_type(4)))  int      v4i;

static constexpr int KD = 4096;
static constexpr int ND = 16384;
static constexpr int MD = 8192;
static constexpr int BM = 256;
static constexpr int BN = 128;
static constexpr int BK = 32;
static constexpr int NK = KD / BK;         // 128 K-steps
static constexpr int LDT = BK + 8;         // padded LDS row stride (halfs): 80B, 16B-aligned
static constexpr int ASZ = BM * LDT;       // 10240 halfs
static constexpr int BSZ = BN * LDT;       //  5120 halfs
static constexpr int BUF = ASZ + BSZ;      // 15360 halfs per ping-pong buffer (60 KB total)

static __device__ inline v8h cvt_f32x8_to_f16(v4f a, v4f b) {
  v8f f = __builtin_shufflevector(a, b, 0, 1, 2, 3, 4, 5, 6, 7);
  return __builtin_convertvector(f, v8h);
}

static __device__ inline v16h frag16(const _Float16* p) {
  v8h lo = *(const v8h*)p;          // K chunk {fk .. fk+7}
  v8h hi = *(const v8h*)(p + 16);   // K chunk {fk+16 .. fk+23}
  return __builtin_shufflevector(lo, hi, 0, 1, 2, 3, 4, 5, 6, 7,
                                         8, 9, 10, 11, 12, 13, 14, 15);
}

// staged (converted) tile data held in registers between global load and LDS store
struct Staged {
  v8h x0, x1, x2, x3;   // 32 halfs: one full x row (BK) per thread
  v8h w0, w1;           // 16 halfs: half a weight row per thread
};

static __device__ inline Staged stage_load(const float* __restrict__ xg,
                                           const int*   __restrict__ wg,
                                           float scale, float nzs) {
  v4f a0 = *(const v4f*)(xg);
  v4f a1 = *(const v4f*)(xg + 4);
  v4f a2 = *(const v4f*)(xg + 8);
  v4f a3 = *(const v4f*)(xg + 12);
  v4f a4 = *(const v4f*)(xg + 16);
  v4f a5 = *(const v4f*)(xg + 20);
  v4f a6 = *(const v4f*)(xg + 24);
  v4f a7 = *(const v4f*)(xg + 28);

  v4i q0 = *(const v4i*)(wg);
  v4i q1 = *(const v4i*)(wg + 4);
  v4i q2 = *(const v4i*)(wg + 8);
  v4i q3 = *(const v4i*)(wg + 12);

  // dequant as fma: q*scale + (-zp*scale)  -> v_pk_fma_f32
  v4f b0 = __builtin_convertvector(q0, v4f) * scale + nzs;
  v4f b1 = __builtin_convertvector(q1, v4f) * scale + nzs;
  v4f b2 = __builtin_convertvector(q2, v4f) * scale + nzs;
  v4f b3 = __builtin_convertvector(q3, v4f) * scale + nzs;

  Staged s;
  s.x0 = cvt_f32x8_to_f16(a0, a1);
  s.x1 = cvt_f32x8_to_f16(a2, a3);
  s.x2 = cvt_f32x8_to_f16(a4, a5);
  s.x3 = cvt_f32x8_to_f16(a6, a7);
  s.w0 = cvt_f32x8_to_f16(b0, b1);
  s.w1 = cvt_f32x8_to_f16(b2, b3);
  return s;
}

__global__ __launch_bounds__(256)
void qlinear_wmma_f16(const float* __restrict__ x,
                      const int*   __restrict__ qw,
                      const float* __restrict__ wscale,
                      const float* __restrict__ wzp,
                      const float* __restrict__ bias,
                      float* __restrict__ out)
{
  // ping-pong: [buf][As(256x40) | Bs(128x40)] halfs; total 60 KB
  __shared__ alignas(16) _Float16 smem[2 * BUF];

  const int tid  = threadIdx.x;
  const int lane = tid & 31;
  const int wave = tid >> 5;

  const int bm0 = blockIdx.x * BM;   // M fastest -> weight panels reused across M blocks in L2
  const int bn0 = blockIdx.y * BN;

  const float scale = wscale[0];
  const float nzs   = -wzp[0] * scale;

  // 8 waves tiled 4 (M) x 2 (N): each wave owns a 64x64 output sub-tile
  const int wm = (wave & 3) * 64;
  const int wn = (wave >> 2) * 64;

  v8f acc[4][4];
#pragma unroll
  for (int i = 0; i < 4; ++i)
#pragma unroll
    for (int j = 0; j < 4; ++j) {
      v8f z = {0.f, 0.f, 0.f, 0.f, 0.f, 0.f, 0.f, 0.f};
      acc[i][j] = z;
    }

  // staging map: x -> one full row (BK) per thread; w -> half row per thread
  const int xrow = tid;                // 0..255
  const int wrow = tid >> 1;           // 0..127
  const int wcol = (tid & 1) << 4;     // 0 or 16

  const float* xg = x  + (size_t)(bm0 + xrow) * KD;
  const int*   wg = qw + (size_t)(bn0 + wrow) * KD + wcol;

  // WMMA 16-bit A/B fragment addressing (ISA 7.12.2):
  // lanes 0-15: row=lane,     K chunks {0..7} and {16..23}
  // lanes 16-31: row=lane-16, K chunks {8..15} and {24..31}
  const int fr = lane & 15;
  const int fk = (lane >> 4) << 3;     // 0 or 8

  // ---- prologue: stage tile 0 into buffer 0
  {
    Staged s = stage_load(xg, wg, scale, nzs);
    _Float16* As = smem;
    _Float16* Bs = smem + ASZ;
    *(v8h*)&As[xrow * LDT]      = s.x0;
    *(v8h*)&As[xrow * LDT + 8]  = s.x1;
    *(v8h*)&As[xrow * LDT + 16] = s.x2;
    *(v8h*)&As[xrow * LDT + 24] = s.x3;
    *(v8h*)&Bs[wrow * LDT + wcol]     = s.w0;
    *(v8h*)&Bs[wrow * LDT + wcol + 8] = s.w1;
  }

  // ---- steady state: single barrier per K-step, ping-pong buffers
  for (int k = 0; k < NK - 1; ++k) {
    __syncthreads();   // buf[k&1] ready; buf[(k+1)&1] free (skew bounded by barrier)

    // issue next tile's global loads early; conversion VALU co-issues with WMMA
    Staged s = stage_load(xg + (k + 1) * BK, wg + (k + 1) * BK, scale, nzs);

    const _Float16* As = smem + (k & 1) * BUF;
    const _Float16* Bs = As + ASZ;

    // B fragments persist across the mt loop (each reused by 4 WMMAs)
    v16h bf[4];
#pragma unroll
    for (int nt = 0; nt < 4; ++nt)
      bf[nt] = frag16(&Bs[(wn + nt * 16 + fr) * LDT + fk]);

#pragma unroll
    for (int mt = 0; mt < 4; ++mt) {
      v16h af = frag16(&As[(wm + mt * 16 + fr) * LDT + fk]);
#pragma unroll
      for (int nt = 0; nt < 4; ++nt)
        acc[mt][nt] = __builtin_amdgcn_wmma_f32_16x16x32_f16(
            false, af, false, bf[nt],
            (short)0, acc[mt][nt], false, false);
    }

    // store next tile into the other buffer (no reader conflicts pre-barrier)
    _Float16* Asn = smem + ((k + 1) & 1) * BUF;
    _Float16* Bsn = Asn + ASZ;
    *(v8h*)&Asn[xrow * LDT]      = s.x0;
    *(v8h*)&Asn[xrow * LDT + 8]  = s.x1;
    *(v8h*)&Asn[xrow * LDT + 16] = s.x2;
    *(v8h*)&Asn[xrow * LDT + 24] = s.x3;
    *(v8h*)&Bsn[wrow * LDT + wcol]     = s.w0;
    *(v8h*)&Bsn[wrow * LDT + wcol + 8] = s.w1;
  }

  // ---- peeled last K-step (no staging)
  {
    __syncthreads();
    const _Float16* As = smem + ((NK - 1) & 1) * BUF;
    const _Float16* Bs = As + ASZ;

    v16h bf[4];
#pragma unroll
    for (int nt = 0; nt < 4; ++nt)
      bf[nt] = frag16(&Bs[(wn + nt * 16 + fr) * LDT + fk]);

#pragma unroll
    for (int mt = 0; mt < 4; ++mt) {
      v16h af = frag16(&As[(wm + mt * 16 + fr) * LDT + fk]);
#pragma unroll
      for (int nt = 0; nt < 4; ++nt)
        acc[mt][nt] = __builtin_amdgcn_wmma_f32_16x16x32_f16(
            false, af, false, bf[nt],
            (short)0, acc[mt][nt], false, false);
    }
  }

  // ---- epilogue: C/D layout -> VGPR i holds M=i (lanes 0-15) / M=i+8 (lanes 16-31), N=lane&15
  const int mofs = (lane >> 4) << 3;
#pragma unroll
  for (int nt = 0; nt < 4; ++nt) {
    const int n = bn0 + wn + nt * 16 + fr;
    const float bv = bias[n];
#pragma unroll
    for (int mt = 0; mt < 4; ++mt) {
      const int m = bm0 + wm + mt * 16 + mofs;
#pragma unroll
      for (int i = 0; i < 8; ++i)
        out[(size_t)(m + i) * ND + n] = acc[mt][nt][i] + bv;
    }
  }
}

extern "C" void kernel_launch(void* const* d_in, const int* in_sizes, int n_in,
                              void* d_out, int out_size, void* d_ws, size_t ws_size,
                              hipStream_t stream) {
  (void)in_sizes; (void)n_in; (void)out_size; (void)d_ws; (void)ws_size;
  const float* x      = (const float*)d_in[0];
  const int*   qw     = (const int*)d_in[1];
  const float* wscale = (const float*)d_in[2];
  const float* wzp    = (const float*)d_in[3];
  const float* bias   = (const float*)d_in[4];
  float*       out    = (float*)d_out;

  dim3 grid(MD / BM, ND / BN);   // (32, 128), M fastest for L2 weight-panel reuse
  dim3 block(256);               // 8 wave32 waves
  qlinear_wmma_f16<<<grid, block, 0, stream>>>(x, qw, wscale, wzp, bias, out);
}
